// GCN_64304250356313
// MI455X (gfx1250) — matrix operand order repaired and checked by reading, not compiled, not run on hardware
//
#include <hip/hip_runtime.h>
#include <hip/hip_bf16.h>
#include <stdint.h>

// ---------------------------------------------------------------------------
// GCN 2-layer forward for MI455X (gfx1250, wave32).
//  - Layer-1 GEMM: v_wmma_f32_16x16x32_bf16, tiles staged by the Tensor Data
//    Mover (tensor_load_to_lds, TENSORcnt double-buffered, TDM pad feature
//    gives 68B LDS row pitch -> conflict-free fragment reads).
//  - Edge aggregation: global_atomic_add_f32 via unsafeAtomicAdd.
// ---------------------------------------------------------------------------

#define IN_C   512
#define HID    128
#define BLK_M  128       // rows per workgroup in GEMM
#define KT     32        // bf16 WMMA K depth
#define NKT    (IN_C / KT)
#define GEMM_THREADS 256 // 8 waves
#define ROW_PITCH 68     // 64B tile row + 4B TDM pad
#define TILE_BYTES (BLK_M * ROW_PITCH)   // 8704 per tile

typedef __attribute__((ext_vector_type(16))) __bf16 v16bf;
typedef __attribute__((ext_vector_type(8)))  float  v8f;
typedef __attribute__((ext_vector_type(4)))  unsigned int v4u;
typedef __attribute__((ext_vector_type(8)))  int v8i;
typedef __attribute__((ext_vector_type(4)))  int v4i;

union AFrag { uint32_t u[8]; v16bf v; };

__device__ __forceinline__ __bf16 f2bf(float f) {
    uint32_t u = __builtin_bit_cast(uint32_t, f);
    uint32_t r = (u + 0x7FFFu + ((u >> 16) & 1u)) >> 16;   // round-to-nearest-even
    uint16_t s = (uint16_t)r;
    return __builtin_bit_cast(__bf16, s);
}

// ---- TDM: DMA one 32(k) x 128(rows) bf16 tile into LDS (ISA 8.3/8.4 D#) ----
// rows beyond rows_rem are zero-filled by TDM OOB handling; pad_interval=3
// (16 DWORDs) + pad_amount=0 (1 DWORD) => LDS row pitch 68 bytes.
__device__ __forceinline__ void tdm_load_tile(uint32_t lds_off, const __bf16* gsrc,
                                              uint32_t dim0_rem, uint32_t rows_rem,
                                              uint32_t stride_elems) {
    uint64_t ga = (uint64_t)(uintptr_t)gsrc;
    v4u g0;
    g0[0] = 1u;                                   // count=1, user descriptor
    g0[1] = lds_off;                              // lds_addr (bytes)
    g0[2] = (uint32_t)ga;                         // global_addr[31:0]
    g0[3] = (uint32_t)((ga >> 32) & 0x01FFFFFFu) | (2u << 30);  // addr[56:32], type=2
    v8i g1;
    g1[0] = (int)((1u << 16) | (1u << 20) | (3u << 22));  // data_size=2B, pad_en, int=16DW, amt=1DW
    g1[1] = (int)(dim0_rem << 16);                        // tensor_dim0[15:0] @ [63:48]
    g1[2] = (int)((dim0_rem >> 16) | ((rows_rem & 0xFFFFu) << 16)); // dim0 hi | dim1 lo
    g1[3] = (int)((rows_rem >> 16) | (32u << 16));        // dim1 hi | tile_dim0=32
    g1[4] = (int)(128u);                                  // tile_dim1=128, tile_dim2=0
    g1[5] = (int)stride_elems;                            // tensor_dim0_stride[31:0]
    g1[6] = 0;                                            // stride hi | dim1_stride lo
    g1[7] = 0;
    v4i z4 = {0, 0, 0, 0};
    v8i z8 = {0, 0, 0, 0, 0, 0, 0, 0};
    // clang-23 / therock-10.0 form: 6 args (g0, g1, g2, g3, extra group, cpol)
    __builtin_amdgcn_tensor_load_to_lds(g0, g1, z4, z4, z8, 0);
}

// ---------------- precision pre-passes (f32 -> bf16) ----------------

__global__ void k_cvt_x(const float* __restrict__ x, __bf16* __restrict__ xb, int total) {
    int i = blockIdx.x * blockDim.x + threadIdx.x;
    if (i < total) xb[i] = f2bf(x[i]);
}

// W1 [k][n] f32  ->  w1t [n][k] bf16 (so B-tile rows are K-contiguous)
__global__ void k_cvt_w1t(const float* __restrict__ W1, __bf16* __restrict__ w1t) {
    int idx = blockIdx.x * blockDim.x + threadIdx.x;
    if (idx >= IN_C * HID) return;
    int nn = idx & (HID - 1), k = idx >> 7;
    w1t[(size_t)nn * IN_C + k] = f2bf(W1[(size_t)k * HID + nn]);
}

// ---------------- degree / normalization ----------------

__global__ void k_deg_init(float* deg, int n) {
    int i = blockIdx.x * blockDim.x + threadIdx.x;
    if (i < n) deg[i] = 1.0f;   // self-loop
}

__global__ void k_deg_accum(const int* col, float* deg, int e) {
    int i = blockIdx.x * blockDim.x + threadIdx.x;
    if (i < e) unsafeAtomicAdd(&deg[col[i]], 1.0f);
}

__global__ void k_rsqrt(float* deg, int n) {
    int i = blockIdx.x * blockDim.x + threadIdx.x;
    if (i < n) deg[i] = __frsqrt_rn(deg[i]);   // deg >= 1 always
}

// ---------------- layer-1 GEMM: h1 = x @ W1 (TDM + bf16 WMMA) ----------

__global__ __launch_bounds__(GEMM_THREADS)
void k_gemm1(const __bf16* __restrict__ xb, const __bf16* __restrict__ w1t,
             float* __restrict__ h1, int nrows) {
    // double-buffered A/B tiles, each 128 rows x 68B pitch (TDM-padded)
    __shared__ __align__(16) char smem[4 * TILE_BYTES];

    const int t      = threadIdx.x;
    const int wave   = t >> 5;            // 0..7 : 16-row strip per wave
    const int lane   = t & 31;
    const int half   = lane >> 4;
    const int l15    = lane & 15;
    const int m0     = blockIdx.x * BLK_M;
    const int mLocal = wave * 16 + l15;

    const uint32_t lds_base = (uint32_t)(uintptr_t)(void*)smem;  // LDS byte offset
    const uint32_t offA[2] = { 0u, 2u * TILE_BYTES };
    const uint32_t offB[2] = { 1u * TILE_BYTES, 3u * TILE_BYTES };
    const uint32_t rows_rem = (uint32_t)(nrows - m0);

    v8f acc[8];
    #pragma unroll
    for (int nt = 0; nt < 8; ++nt) acc[nt] = (v8f){0,0,0,0,0,0,0,0};

    // prologue: prefetch k-steps 0 and 1
    if (wave == 0) {
        #pragma unroll
        for (int p = 0; p < 2; ++p) {
            int k0 = p * KT;
            tdm_load_tile(lds_base + offA[p], xb  + (size_t)m0 * IN_C + k0,
                          (uint32_t)(IN_C - k0), rows_rem, IN_C);
            tdm_load_tile(lds_base + offB[p], w1t + k0,
                          (uint32_t)(IN_C - k0), (uint32_t)HID, IN_C);
        }
    }

    for (int kt = 0; kt < NKT; ++kt) {
        const int buf = kt & 1;
        if (wave == 0) {
            if (kt + 1 < NKT) __builtin_amdgcn_s_wait_tensorcnt(2);  // oldest A+B pair done
            else              __builtin_amdgcn_s_wait_tensorcnt(0);
        }
        __syncthreads();

        // ---- A fragment (16x32 bf16, ISA 7.12.2 layout), 68B row pitch
        AFrag af;
        const uint32_t* rowA = (const uint32_t*)(smem + offA[buf] + mLocal * ROW_PITCH);
        #pragma unroll
        for (int j = 0; j < 8; ++j) {
            int kk = (j < 4) ? (8 * half + 2 * j) : (16 + 8 * half + 2 * (j - 4));
            af.u[j] = rowA[kk >> 1];
        }
        // ---- 8 column tiles: acc[nt] += A * B[:, nt*16+l15]
        #pragma unroll
        for (int nt = 0; nt < 8; ++nt) {
            AFrag bf_;
            const uint32_t* rowB =
                (const uint32_t*)(smem + offB[buf] + (nt * 16 + l15) * ROW_PITCH);
            #pragma unroll
            for (int j = 0; j < 8; ++j) bf_.u[j] = rowB[4 * half + j + 4 * (j >> 2)];  // K=8*half+2j (j<4), 16+8*half+2(j-4) (j>=4)
            acc[nt] = __builtin_amdgcn_wmma_f32_16x16x32_bf16(
                false, af.v, false, bf_.v, (short)0, acc[nt], false, false);
        }
        __syncthreads();   // all waves done reading buf before DMA overwrites it

        if (wave == 0 && kt + 2 < NKT) {
            int k0 = (kt + 2) * KT;
            tdm_load_tile(lds_base + offA[buf], xb  + (size_t)m0 * IN_C + k0,
                          (uint32_t)(IN_C - k0), rows_rem, IN_C);
            tdm_load_tile(lds_base + offB[buf], w1t + k0,
                          (uint32_t)(IN_C - k0), (uint32_t)HID, IN_C);
        }
    }

    // ---- store D: VGPR r -> M = 8*half + r within tile, N = l15
    #pragma unroll
    for (int nt = 0; nt < 8; ++nt) {
        int col = nt * 16 + l15;
        #pragma unroll
        for (int r = 0; r < 8; ++r) {
            int row = m0 + wave * 16 + 8 * half + r;
            if (row < nrows) h1[(size_t)row * HID + col] = acc[nt][r];
        }
    }
}

// ---------------- layer-1 aggregation ----------------

// agg[i][c] = b1[c] + h1[i][c] * dinv[i]^2   (bias + self-loop folded)
__global__ void k_agg_init(const float* h1, const float* b1, const float* dinv,
                           float* agg, int n) {
    int idx = blockIdx.x * blockDim.x + threadIdx.x;
    if (idx >= n * HID) return;
    int i = idx >> 7, c = idx & 127;
    float d = dinv[i];
    agg[idx] = b1[c] + h1[idx] * d * d;
}

// one wave per edge; 128 channels in 4 strided passes
__global__ void k_agg_edges(const int* edge, const float* h1, const float* dinv,
                            float* agg, int e) {
    int warp = (blockIdx.x * blockDim.x + threadIdx.x) >> 5;
    int lane = threadIdx.x & 31;
    if (warp >= e) return;
    int row = edge[warp];
    int col = edge[warp + e];
    float s = dinv[row] * dinv[col];
    const float* src = h1 + (size_t)row * HID;
    float*       dst = agg + (size_t)col * HID;
    #pragma unroll
    for (int p = 0; p < 4; ++p) {
        int c = lane + p * 32;
        unsafeAtomicAdd(&dst[c], src[c] * s);
    }
}

// ---------------- layer 2: z[i] = relu(agg[i]) . W2 ; out[i] = b2 + z[i]*dinv^2
__global__ void k_layer2_node(const float* agg, const float* W2, const float* b2,
                              const float* dinv, float* z, float* out, int n) {
    int warp = (blockIdx.x * blockDim.x + threadIdx.x) >> 5;
    int lane = threadIdx.x & 31;
    if (warp >= n) return;
    const float* a = agg + (size_t)warp * HID;
    float acc = 0.0f;
    #pragma unroll
    for (int p = 0; p < 4; ++p) {
        int c = lane + p * 32;
        float v = a[c];
        acc += fmaxf(v, 0.0f) * W2[c];
    }
    #pragma unroll
    for (int off = 16; off > 0; off >>= 1)
        acc += __shfl_xor(acc, off, 32);
    if (lane == 0) {
        float d = dinv[warp];
        z[warp]   = acc;
        out[warp] = b2[0] + acc * d * d;
    }
}

__global__ void k_scatter2(const int* edge, const float* z, const float* dinv,
                           float* out, int e) {
    int i = blockIdx.x * blockDim.x + threadIdx.x;
    if (i >= e) return;
    int row = edge[i];
    int col = edge[i + e];
    unsafeAtomicAdd(&out[col], z[row] * dinv[row] * dinv[col]);
}

__global__ void k_sigmoid(float* out, int n) {
    int i = blockIdx.x * blockDim.x + threadIdx.x;
    if (i < n) out[i] = 1.0f / (1.0f + __expf(-out[i]));
}

// ---------------------------------------------------------------------------

extern "C" void kernel_launch(void* const* d_in, const int* in_sizes, int n_in,
                              void* d_out, int out_size, void* d_ws, size_t ws_size,
                              hipStream_t stream) {
    const float* x    = (const float*)d_in[0];
    const float* W1   = (const float*)d_in[1];
    const float* b1   = (const float*)d_in[2];
    const float* W2   = (const float*)d_in[3];
    const float* b2   = (const float*)d_in[4];
    const int*   edge = (const int*)d_in[5];   // [2, E] int32
    float*       out  = (float*)d_out;

    const int n = in_sizes[0] / IN_C;          // 50000
    const int e = in_sizes[5] / 2;             // 800000
    const int npad = ((n + BLK_M - 1) / BLK_M) * BLK_M;

    // workspace layout: dinv | h1 | agg | z | xb(bf16) | w1t(bf16)  (~104 MB)
    char* base = (char*)d_ws;
    size_t off = 0;
    float* dinv = (float*)(base + off); off += (size_t)npad * 4;
    float* h1   = (float*)(base + off); off += (size_t)npad * HID * 4;
    float* agg  = (float*)(base + off); off += (size_t)npad * HID * 4;
    float* z    = (float*)(base + off); off += (size_t)npad * 4;
    __bf16* xb  = (__bf16*)(base + off); off += (size_t)n * IN_C * 2;
    __bf16* w1t = (__bf16*)(base + off); off += (size_t)HID * IN_C * 2;

    const int T = 256;
    // bf16 pre-conversion for the TDM/WMMA GEMM
    k_cvt_x  <<<((size_t)n * IN_C + T - 1) / T, T, 0, stream>>>(x, xb, n * IN_C);
    k_cvt_w1t<<<(IN_C * HID + T - 1) / T, T, 0, stream>>>(W1, w1t);

    // degree (with self-loops) -> dinv = rsqrt(deg)
    k_deg_init <<<(n + T - 1) / T, T, 0, stream>>>(dinv, n);
    k_deg_accum<<<(e + T - 1) / T, T, 0, stream>>>(edge + e, dinv, e);
    k_rsqrt    <<<(n + T - 1) / T, T, 0, stream>>>(dinv, n);

    // h1 = x @ W1  (TDM-staged bf16 WMMA)
    k_gemm1<<<npad / BLK_M, GEMM_THREADS, 0, stream>>>(xb, w1t, h1, n);

    // agg = segment_sum(h1[row]*norm, col) + b1   (self-loop folded into init)
    k_agg_init <<<((size_t)n * HID + T - 1) / T, T, 0, stream>>>(h1, b1, dinv, agg, n);
    k_agg_edges<<<((size_t)e * 32 + T - 1) / T, T, 0, stream>>>(edge, h1, dinv, agg, e);

    // layer 2: relu -> GEMV -> aggregate -> sigmoid
    k_layer2_node<<<((size_t)n * 32 + T - 1) / T, T, 0, stream>>>(agg, W2, b2, dinv, z, out, n);
    k_scatter2   <<<(e + T - 1) / T, T, 0, stream>>>(edge, z, dinv, out, e);
    k_sigmoid    <<<(n + T - 1) / T, T, 0, stream>>>(out, n);
}
